// MultiHeadLatentAttention_704374636975
// MI455X (gfx1250) — compile-verified, matrix operand order checked by reading, not run on hardware
//
#include <hip/hip_runtime.h>
#include <hip/hip_bf16.h>
#include <math.h>

// ---------------------------------------------------------------------------
// MLA (Multi-Head Latent Attention) for gfx1250 — WMMA f16->f32 everywhere.
// B=4 S=2048 DM=1024 DE=1024 H=16 DH=64 DC=128 DCq=128 DR=32, d_qk = 96.
// ---------------------------------------------------------------------------

#define B_   4
#define S_   2048
#define DM_  1024
#define DE_  1024
#define H_   16
#define DH_  64
#define DC_  128
#define DCQ_ 128
#define DR_  32
#define DQK_ 96
#define M_TOT (B_ * S_)   // 8192 token rows

typedef __attribute__((ext_vector_type(16))) _Float16 v16h;
typedef __attribute__((ext_vector_type(8)))  float    v8f;

__device__ __constant__ float kScale = 0.10206207261596575f; // 1/sqrt(96)

// ---------------------------------------------------------------------------
// Elementwise fp32 -> f16
// ---------------------------------------------------------------------------
__global__ __launch_bounds__(256) void cvt_f32_to_f16(const float* __restrict__ src,
                                                      _Float16* __restrict__ dst,
                                                      size_t n) {
  size_t i = (size_t)blockIdx.x * blockDim.x + threadIdx.x;
  if (i < n) dst[i] = (_Float16)src[i];
}

// Transpose + convert: W[din][dout] (fp32, row-major) -> Wt[dout][din] (f16)
__global__ __launch_bounds__(256) void transconv_f16(const float* __restrict__ W,
                                                     _Float16* __restrict__ Wt,
                                                     int din, int dout) {
  int i = blockIdx.x * blockDim.x + threadIdx.x;
  int total = din * dout;
  if (i >= total) return;
  int n = i / din;          // output feature
  int k = i - n * din;      // input feature
  Wt[i] = (_Float16)W[(size_t)k * dout + n];
}

// ---------------------------------------------------------------------------
// Generic WMMA GEMM:  C[M][N] = A[M][K] * Bt[N][K]^T + bias[N]
// A row-major f16, Bt is B transposed (N-major) f16 so the per-lane
// B-fragment (column-striped) is a contiguous 32B load.
// Block = 256 threads = 8 waves; wave w owns a 16x16 tile; block covers
// 128 rows x 16 cols.  Grid = (M/128, N/16).
// ---------------------------------------------------------------------------
template <typename OutT>
__global__ __launch_bounds__(256) void gemm_wmma_f16(const _Float16* __restrict__ A,
                                                     const _Float16* __restrict__ Bt,
                                                     const float* __restrict__ bias,
                                                     OutT* __restrict__ C,
                                                     int M, int N, int K) {
  const int wave = threadIdx.x >> 5;
  const int lane = threadIdx.x & 31;
  const int m0 = (blockIdx.x * 8 + wave) * 16;
  const int n0 = blockIdx.y * 16;
  const int lrow = lane & 15;        // A row / B column within tile
  const int lsel = lane >> 4;        // which K-half this lane holds

  const _Float16* arow = A  + (size_t)(m0 + lrow) * K + (lsel << 4);
  const _Float16* brow = Bt + (size_t)(n0 + lrow) * K + (lsel << 4);

  v8f acc = {};
  for (int kk = 0; kk < K; kk += 32) {
    // speculative prefetch 8 K-chunks ahead (global_prefetch_b8)
    __builtin_prefetch(arow + kk + 256, 0, 1);
    __builtin_prefetch(brow + kk + 256, 0, 1);
    v16h a = *(const v16h*)(arow + kk);
    v16h b = *(const v16h*)(brow + kk);
    acc = __builtin_amdgcn_wmma_f32_16x16x32_f16(false, a, false, b,
                                                 (short)0, acc, false, false);
  }

  const int rbase = lsel << 3;           // C rows: lanes 0-15 -> M=r, 16-31 -> M=8+r
  const int col   = n0 + lrow;
  const float bv  = bias ? bias[col] : 0.0f;
#pragma unroll
  for (int r = 0; r < 8; ++r) {
    C[(size_t)(m0 + rbase + r) * N + col] = (OutT)(acc[r] + bv);
  }
}

// ---------------------------------------------------------------------------
// Assemble q_t/k_t (RoPE applied to the last 32 dims) and V (d-major).
//   q_t,k_t : [B][H][S][96]   f16
//   v_t     : [B][H][64][S]   f16  (transposed so PV B-fragments are contiguous)
// One thread per (token m, head h).
// ---------------------------------------------------------------------------
__global__ __launch_bounds__(256) void mla_assemble(const _Float16* __restrict__ qc,  // [M][1024]
                                                    const _Float16* __restrict__ kc,  // [M][1024]
                                                    const _Float16* __restrict__ vc,  // [M][1024]
                                                    const _Float16* __restrict__ qr,  // [M][512]
                                                    const _Float16* __restrict__ kr,  // [M][32]
                                                    _Float16* __restrict__ q_t,
                                                    _Float16* __restrict__ k_t,
                                                    _Float16* __restrict__ v_t) {
  int idx = blockIdx.x * blockDim.x + threadIdx.x;
  if (idx >= M_TOT * H_) return;
  const int h = idx & (H_ - 1);
  const int m = idx >> 4;
  const int b = m / S_;
  const int s = m - b * S_;

  const size_t bh   = (size_t)b * H_ + h;
  const size_t base = (bh * S_ + s) * DQK_;

#pragma unroll 8
  for (int d = 0; d < DH_; ++d) {
    q_t[base + d] = qc[(size_t)m * DE_ + h * DH_ + d];
    k_t[base + d] = kc[(size_t)m * DE_ + h * DH_ + d];
  }

  // RoPE: inv_freq_j = 10000^(-j/16), j = 0..15
  const float ln10k_over16 = 0.5756462732485114f;  // ln(10000)/16
#pragma unroll
  for (int j = 0; j < DR_ / 2; ++j) {
    float ang = (float)s * __expf(-(float)j * ln10k_over16);
    float c, sn;
    __sincosf(ang, &sn, &c);
    float qrr = (float)qr[(size_t)m * (H_ * DR_) + h * DR_ + 2 * j];
    float qri = (float)qr[(size_t)m * (H_ * DR_) + h * DR_ + 2 * j + 1];
    q_t[base + DH_ + 2 * j]     = (_Float16)(qrr * c - qri * sn);
    q_t[base + DH_ + 2 * j + 1] = (_Float16)(qrr * sn + qri * c);
    float krr = (float)kr[(size_t)m * DR_ + 2 * j];
    float kri = (float)kr[(size_t)m * DR_ + 2 * j + 1];
    k_t[base + DH_ + 2 * j]     = (_Float16)(krr * c - kri * sn);
    k_t[base + DH_ + 2 * j + 1] = (_Float16)(krr * sn + kri * c);
  }

  const size_t vb = (bh * DH_) * (size_t)S_ + s;
#pragma unroll 8
  for (int d = 0; d < DH_; ++d)
    v_t[vb + (size_t)d * S_] = vc[(size_t)m * DE_ + h * DH_ + d];
}

// ---------------------------------------------------------------------------
// Attention: one block per (b, h, 16-query tile).  128 threads = 4 waves.
//  Phase 1: each wave computes causal score tiles (3x WMMA over d=96) and
//           stores scaled/masked scores as f16 into a 16 x S LDS slab
//           (64 KB of CDNA5's 320KB/WGP LDS -> several WGs per WGP).
//  Phase 2: softmax in LDS (8 threads/row, __shfl_xor width-8 reductions);
//           exp values written back UNNORMALIZED as f16; 1/rowsum kept in
//           s_inv[] and folded into the PV accumulator (normalization is
//           per query row == per PV output row, so it commutes with PV).
//  Phase 3: P·V via WMMA; P fragments are direct 32B vector LDS loads
//           (ds_load_b128), no per-element rebuild.  Output f16 [B][S][1024].
// ---------------------------------------------------------------------------
__global__ __launch_bounds__(128) void mla_attention(const _Float16* __restrict__ q_t,
                                                     const _Float16* __restrict__ k_t,
                                                     const _Float16* __restrict__ v_t,
                                                     _Float16* __restrict__ attn_out) {
  extern __shared__ _Float16 sc_h[];     // [16][S_] f16: raw scores -> exp values
  __shared__ float s_inv[16];

  const int qt_i = blockIdx.x;
  const int h    = blockIdx.y;
  const int b    = blockIdx.z;
  const int tid  = threadIdx.x;
  const int wave = tid >> 5;
  const int lane = tid & 31;
  const int lrow = lane & 15;
  const int lsel = lane >> 4;
  const int rbase = lsel << 3;
  const int q0 = qt_i * 16;

  const size_t bh = (size_t)b * H_ + h;

  // Q fragments for this wave's 16 query rows (3 K-chunks of 32 over d=96)
  const _Float16* qrow = q_t + (bh * S_ + q0 + lrow) * DQK_ + (lsel << 4);
  v16h qa0 = *(const v16h*)(qrow);
  v16h qa1 = *(const v16h*)(qrow + 32);
  v16h qa2 = *(const v16h*)(qrow + 64);

  const int ktiles = qt_i + 1;           // causal: key tiles 0..qt_i
  for (int kt_i = wave; kt_i < ktiles; kt_i += 4) {
    const _Float16* krow = k_t + (bh * S_ + kt_i * 16 + lrow) * DQK_ + (lsel << 4);
    __builtin_prefetch(krow + 64 * DQK_, 0, 1);   // next tile for this wave
    v16h kb0 = *(const v16h*)(krow);
    v16h kb1 = *(const v16h*)(krow + 32);
    v16h kb2 = *(const v16h*)(krow + 64);
    v8f acc = {};
    acc = __builtin_amdgcn_wmma_f32_16x16x32_f16(false, qa0, false, kb0, (short)0, acc, false, false);
    acc = __builtin_amdgcn_wmma_f32_16x16x32_f16(false, qa1, false, kb1, (short)0, acc, false, false);
    acc = __builtin_amdgcn_wmma_f32_16x16x32_f16(false, qa2, false, kb2, (short)0, acc, false, false);

    const int key = kt_i * 16 + lrow;    // this lane's column (key index)
#pragma unroll
    for (int r = 0; r < 8; ++r) {
      const int q = q0 + rbase + r;
      float v = acc[r] * kScale;
      if (key > q) v = -INFINITY;        // causal mask
      sc_h[(rbase + r) * S_ + key] = (_Float16)v;
    }
  }
  __syncthreads();

  // ---- softmax: 16 rows x 8 threads; exp written back unnormalized ----
  {
    const int row = tid >> 3;
    const int sub = tid & 7;
    const int L = ktiles * 16;
    float m = -INFINITY;
    for (int c = sub; c < L; c += 8) m = fmaxf(m, (float)sc_h[row * S_ + c]);
#pragma unroll
    for (int off = 4; off > 0; off >>= 1) m = fmaxf(m, __shfl_xor(m, off, 8));
    float sum = 0.0f;
    for (int c = sub; c < L; c += 8) {
      float e = __expf((float)sc_h[row * S_ + c] - m);
      sc_h[row * S_ + c] = (_Float16)e;
      sum += e;
    }
#pragma unroll
    for (int off = 4; off > 0; off >>= 1) sum += __shfl_xor(sum, off, 8);
    if (sub == 0) s_inv[row] = 1.0f / sum;
    // zero-pad P up to a multiple of 32 keys for the WMMA K-loop
    const int Lpad = (L + 31) & ~31;
    if (Lpad > L) {
      for (int e = tid; e < 16 * 16; e += 128)
        sc_h[(e >> 4) * S_ + L + (e & 15)] = (_Float16)0.0f;
    }
  }
  __syncthreads();

  // ---- P·V: A-fragments are straight 32B LDS vector loads ----
  {
    const int d0 = wave * 16;            // 4 waves x 16 dims = 64
    const int L = ktiles * 16;
    const int Lpad = (L + 31) & ~31;
    const _Float16* vrow = v_t + (bh * DH_ + d0 + lrow) * (size_t)S_ + (lsel << 4);
    const _Float16* prow = sc_h + lrow * S_ + (lsel << 4);
    v8f acc = {};
    for (int kk = 0; kk < Lpad; kk += 32) {
      __builtin_prefetch(vrow + kk + 256, 0, 1);
      v16h pa = *(const v16h*)(prow + kk);
      v16h vb = *(const v16h*)(vrow + kk);
      acc = __builtin_amdgcn_wmma_f32_16x16x32_f16(false, pa, false, vb,
                                                   (short)0, acc, false, false);
    }
#pragma unroll
    for (int r = 0; r < 8; ++r) {
      const int q = q0 + rbase + r;
      const float o = acc[r] * s_inv[rbase + r];   // fold softmax normalization
      attn_out[((size_t)b * S_ + q) * DE_ + h * DH_ + d0 + lrow] = (_Float16)o;
    }
  }
}

// ---------------------------------------------------------------------------
// Host orchestration
// ---------------------------------------------------------------------------
extern "C" void kernel_launch(void* const* d_in, const int* in_sizes, int n_in,
                              void* d_out, int out_size, void* d_ws, size_t ws_size,
                              hipStream_t stream) {
  (void)in_sizes; (void)n_in; (void)out_size; (void)ws_size;

  const float* x     = (const float*)d_in[0];
  // d_in[1] = mask (unused: causal structure is known)
  const float* W_dkv = (const float*)d_in[2];
  const float* b_dkv = (const float*)d_in[3];
  const float* W_dq  = (const float*)d_in[4];
  const float* b_dq  = (const float*)d_in[5];
  const float* W_uk  = (const float*)d_in[6];
  const float* b_uk  = (const float*)d_in[7];
  const float* W_uv  = (const float*)d_in[8];
  const float* b_uv  = (const float*)d_in[9];
  const float* W_uq  = (const float*)d_in[10];
  const float* b_uq  = (const float*)d_in[11];
  const float* W_qr  = (const float*)d_in[12];
  const float* b_qr  = (const float*)d_in[13];
  const float* W_kr  = (const float*)d_in[14];
  const float* b_kr  = (const float*)d_in[15];
  const float* W_o   = (const float*)d_in[16];
  const float* b_o   = (const float*)d_in[17];
  float* out = (float*)d_out;

  size_t off = 0;
  auto halloc = [&](size_t elems) -> _Float16* {
    _Float16* p = (_Float16*)((char*)d_ws + off);
    off += (elems * sizeof(_Float16) + 255) & ~(size_t)255;
    return p;
  };

  _Float16* x_h    = halloc((size_t)M_TOT * DM_);
  _Float16* wt_dq  = halloc((size_t)DCQ_ * DM_);
  _Float16* wt_dkv = halloc((size_t)DC_ * DM_);
  _Float16* wt_kr  = halloc((size_t)DR_ * DM_);
  _Float16* wt_uq  = halloc((size_t)DE_ * DCQ_);
  _Float16* wt_uk  = halloc((size_t)DE_ * DC_);
  _Float16* wt_uv  = halloc((size_t)DE_ * DC_);
  _Float16* wt_qr  = halloc((size_t)(H_ * DR_) * DCQ_);
  _Float16* wt_o   = halloc((size_t)DM_ * DE_);
  _Float16* cq_h   = halloc((size_t)M_TOT * DCQ_);
  _Float16* ckv_h  = halloc((size_t)M_TOT * DC_);
  _Float16* qc_f   = halloc((size_t)M_TOT * DE_);
  _Float16* kc_f   = halloc((size_t)M_TOT * DE_);
  _Float16* vc_f   = halloc((size_t)M_TOT * DE_);
  _Float16* qr_f   = halloc((size_t)M_TOT * (H_ * DR_));
  _Float16* kr_f   = halloc((size_t)M_TOT * DR_);
  _Float16* q_t    = halloc((size_t)B_ * H_ * S_ * DQK_);
  _Float16* k_t    = halloc((size_t)B_ * H_ * S_ * DQK_);
  _Float16* v_t    = halloc((size_t)B_ * H_ * DH_ * S_);
  _Float16* attn_h = halloc((size_t)M_TOT * DE_);

  // 1) conversions
  {
    size_t n = (size_t)M_TOT * DM_;
    cvt_f32_to_f16<<<(unsigned)((n + 255) / 256), 256, 0, stream>>>(x, x_h, n);
  }
  auto tc = [&](const float* W, _Float16* Wt, int din, int dout) {
    int total = din * dout;
    transconv_f16<<<(total + 255) / 256, 256, 0, stream>>>(W, Wt, din, dout);
  };
  tc(W_dq,  wt_dq,  DM_,  DCQ_);
  tc(W_dkv, wt_dkv, DM_,  DC_);
  tc(W_kr,  wt_kr,  DM_,  DR_);
  tc(W_uq,  wt_uq,  DCQ_, DE_);
  tc(W_uk,  wt_uk,  DC_,  DE_);
  tc(W_uv,  wt_uv,  DC_,  DE_);
  tc(W_qr,  wt_qr,  DCQ_, H_ * DR_);
  tc(W_o,   wt_o,   DE_,  DM_);

  // 2) down-projections (K = 1024)
  gemm_wmma_f16<_Float16><<<dim3(M_TOT / 128, DCQ_ / 16), 256, 0, stream>>>(
      x_h, wt_dq, b_dq, cq_h, M_TOT, DCQ_, DM_);
  gemm_wmma_f16<_Float16><<<dim3(M_TOT / 128, DC_ / 16), 256, 0, stream>>>(
      x_h, wt_dkv, b_dkv, ckv_h, M_TOT, DC_, DM_);
  gemm_wmma_f16<_Float16><<<dim3(M_TOT / 128, DR_ / 16), 256, 0, stream>>>(
      x_h, wt_kr, b_kr, kr_f, M_TOT, DR_, DM_);

  // 3) up-projections (K = 128)
  gemm_wmma_f16<_Float16><<<dim3(M_TOT / 128, DE_ / 16), 256, 0, stream>>>(
      cq_h, wt_uq, b_uq, qc_f, M_TOT, DE_, DCQ_);
  gemm_wmma_f16<_Float16><<<dim3(M_TOT / 128, DE_ / 16), 256, 0, stream>>>(
      ckv_h, wt_uk, b_uk, kc_f, M_TOT, DE_, DC_);
  gemm_wmma_f16<_Float16><<<dim3(M_TOT / 128, DE_ / 16), 256, 0, stream>>>(
      ckv_h, wt_uv, b_uv, vc_f, M_TOT, DE_, DC_);
  gemm_wmma_f16<_Float16><<<dim3(M_TOT / 128, (H_ * DR_) / 16), 256, 0, stream>>>(
      cq_h, wt_qr, b_qr, qr_f, M_TOT, H_ * DR_, DCQ_);

  // 4) RoPE + layout assembly
  {
    int total = M_TOT * H_;
    mla_assemble<<<(total + 255) / 256, 256, 0, stream>>>(
        qc_f, kc_f, vc_f, qr_f, kr_f, q_t, k_t, v_t);
  }

  // 5) attention: (S/16) x H x B blocks, 128 threads, 64 KB dynamic LDS
  {
    dim3 g(S_ / 16, H_, B_);
    size_t lds = (size_t)16 * S_ * sizeof(_Float16);
    mla_attention<<<g, 128, lds, stream>>>(q_t, k_t, v_t, attn_h);
  }

  // 6) output projection -> fp32 d_out
  gemm_wmma_f16<float><<<dim3(M_TOT / 128, DM_ / 16), 256, 0, stream>>>(
      attn_h, wt_o, b_o, out, M_TOT, DM_, DE_);
}
